// DyGraphConv2d_43413529428129
// MI455X (gfx1250) — matrix operand order repaired and checked by reading, not compiled, not run on hardware
//
#include <hip/hip_runtime.h>
#include <hip/hip_bf16.h>
#include <math.h>
#include <stdint.h>

// ---------------------------------------------------------------------------
// DyGraphConv2d for MI455X (gfx1250, wave32, WMMA + async Global->LDS).
//
//   h[b,n,k,:] = gelu( X[b,n]·(W1-W2)^T + X[b,nn[b,n,k]]·W2^T + bias )
//   => A1 = X·(W1-W2)^T, A2 = X·W2^T  (bf16 WMMA GEMMs, f32 accumulate),
//      then cheap gather/add/gelu/max over K=9.
// KNN: per-batch gram GEMM fused with register-resident top-9; the 315MB
// distance matrix only ever exists as 64x64 LDS tiles. Xm column tiles are
// streamed with double-buffered GLOBAL_LOAD_ASYNC_TO_LDS_B128 (ASYNCcnt),
// overlapping DMA with WMMA + the top-k scan.
// ---------------------------------------------------------------------------

typedef __bf16 bf16_t;
typedef __bf16 v8bf  __attribute__((ext_vector_type(8)));
typedef __bf16 v16bf __attribute__((ext_vector_type(16)));
typedef float  v8f   __attribute__((ext_vector_type(8)));

#define C_IN   192
#define C_CAT  384
#define O_OUT  192
#define BATCH  8
#define HIN    112
#define HD     56
#define NPTS   3136       // 56*56
#define KNN    9
#define TILE   64
#define NT     49         // 3136/64
#define ROWS_TOT 25088    // 8*3136
#define LDSPITCH 200      // bf16 elems per LDS row (400B, 16B-aligned, odd 16B-granule stride)

#define ASYNC_WAIT_0 asm volatile("s_wait_asynccnt 0x0" ::: "memory")
#define ASYNC_WAIT_6 asm volatile("s_wait_asynccnt 0x6" ::: "memory")

// ---- async DMA tile fill: 64 rows x 192 bf16, 256 threads, 6 B128/thread --
// Per-lane Global->LDS DMA; LDS operand is the low 32 bits of the generic
// shared address.
__device__ __forceinline__ void async_fill_tile(bf16_t (*dst)[LDSPITCH],
                                                const bf16_t* __restrict__ src,
                                                int t) {
  const int row = t >> 2;  // 0..63
  const int seg = t & 3;   // 4 segments of 48 bf16 (96B)
  unsigned long long g =
      (unsigned long long)(uintptr_t)(src + row * C_IN + seg * 48);
  unsigned int l = (unsigned int)(uintptr_t)(&dst[row][seg * 48]);
#pragma unroll
  for (int i = 0; i < 6; ++i) {
    asm volatile("global_load_async_to_lds_b128 %0, %1, off" ::"v"(l), "v"(g)
                 : "memory");
    l += 16;
    g += 16;
  }
}

// ---- WMMA fragment builders (CDNA5 16-bit A 16x32 / B 32x16 layouts) ------
// A: lane<16 -> elems 0..7 = K[kb..kb+7],   elems 8..15 = K[kb+16..kb+23]
//    lane>=16-> elems 0..7 = K[kb+8..kb+15],elems 8..15 = K[kb+24..kb+31]
__device__ __forceinline__ v16bf frag_a(const bf16_t (*X)[LDSPITCH], int row,
                                        int kb, int hi) {
  v8bf lo = *reinterpret_cast<const v8bf*>(&X[row][kb + 8 * hi]);
  v8bf hh = *reinterpret_cast<const v8bf*>(&X[row][kb + 16 + 8 * hi]);
  return __builtin_shufflevector(lo, hh, 0, 1, 2, 3, 4, 5, 6, 7, 8, 9, 10, 11,
                                 12, 13, 14, 15);
}
// B: same K layout as A (lane<16: K[kb..kb+7],K[kb+16..kb+23]; lane>=16: +8)
__device__ __forceinline__ v16bf frag_b(const bf16_t (*X)[LDSPITCH], int row,
                                        int kb, int hi) {
  v8bf lo = *reinterpret_cast<const v8bf*>(&X[row][kb + 8 * hi]);
  v8bf hh = *reinterpret_cast<const v8bf*>(&X[row][kb + 16 + 8 * hi]);
  return __builtin_shufflevector(lo, hh, 0, 1, 2, 3, 4, 5, 6, 7, 8, 9, 10, 11,
                                 12, 13, 14, 15);
}

// ---------------------------------------------------------------------------
// Kernel A: 2x2 avgpool, transpose to X[b][n][c] (bf16) + row sumsq (f32)
// ---------------------------------------------------------------------------
__global__ __launch_bounds__(256) void avgpool_kernel(
    const float* __restrict__ x, bf16_t* __restrict__ xth,
    float* __restrict__ sq) {
  const int b = blockIdx.y;
  const int i = blockIdx.x;  // pooled row 0..55
  const int t = threadIdx.x;
  __shared__ float sT[HD][C_IN];  // 43KB

  for (int idx = t; idx < HD * C_IN; idx += 256) {
    const int j = idx % HD;  // fastest -> coalesced x reads
    const int c = idx / HD;
    const float* p = x + (((size_t)(b * C_IN + c) * HIN) + 2 * i) * HIN + 2 * j;
    sT[j][c] = 0.25f * (p[0] + p[1] + p[HIN] + p[HIN + 1]);
  }
  __syncthreads();
  for (int idx = t; idx < HD * C_IN; idx += 256) {
    const int j = idx / C_IN;
    const int c = idx % C_IN;  // fastest -> coalesced bf16 writes
    xth[((size_t)b * NPTS + i * HD + j) * C_IN + c] = (bf16_t)sT[j][c];
  }
  if (t < HD) {
    float s = 0.f;
#pragma unroll 4
    for (int c = 0; c < C_IN; ++c) {
      float v = sT[t][c];
      s += v * v;
    }
    sq[(size_t)b * NPTS + i * HD + t] = s;
  }
}

// ---------------------------------------------------------------------------
// Kernel P: weight prep  WA = W1 - W2, WB = W2   (row-major [o][c], bf16)
// ---------------------------------------------------------------------------
__global__ __launch_bounds__(256) void prep_w_kernel(
    const float* __restrict__ w, bf16_t* __restrict__ wa,
    bf16_t* __restrict__ wb) {
  const int idx = blockIdx.x * 256 + threadIdx.x;
  if (idx < O_OUT * C_IN) {
    const int o = idx / C_IN, c = idx % C_IN;
    const float w1 = w[o * C_CAT + c];
    const float w2 = w[o * C_CAT + C_IN + c];
    wa[idx] = (bf16_t)(w1 - w2);
    wb[idx] = (bf16_t)w2;
  }
}

// ---------------------------------------------------------------------------
// Kernel B: fused gram GEMM + top-9, double-buffered async Xm stream.
// Ranking key is sq[m] - 2*dot (sq[n] constant per row, dropped). The dist
// tile aliases the *current* Xm buffer once the k-loop is done, while the
// *other* buffer is concurrently DMA-filled with tile mt+1.
// ---------------------------------------------------------------------------
__global__ __launch_bounds__(256) void gram_topk_kernel(
    const bf16_t* __restrict__ xth, const float* __restrict__ sq,
    int* __restrict__ nn) {
  const int b = blockIdx.y;
  const int nbase = blockIdx.x * TILE;
  const int t = threadIdx.x;

  __shared__ __align__(16) bf16_t sXn[TILE][LDSPITCH];     // 25.6KB
  __shared__ __align__(16) bf16_t sXm[2][TILE][LDSPITCH];  // 51.2KB

  const bf16_t* xb = xth + (size_t)b * NPTS * C_IN;

  // prologue: DMA Xn, tile0 -> buf0, tile1 -> buf1; wait for first 12 ops
  async_fill_tile(sXn, xb + (size_t)nbase * C_IN, t);
  async_fill_tile(sXm[0], xb, t);
  async_fill_tile(sXm[1], xb + (size_t)TILE * C_IN, t);
  ASYNC_WAIT_6;  // in-order: Xn + tile0 complete, tile1 may still fly
  __syncthreads();

  const int wv = t >> 5, lane = t & 31;
  const int hi = lane >> 4, nl = lane & 15;
  const int tr = wv & 3;                   // 16-row subtile
  const int tc0 = wv >> 2, tc1 = tc0 + 2;  // two 16-col subtiles

  float bd[KNN];
  int bi[KNN];
#pragma unroll
  for (int k = 0; k < KNN; ++k) {
    bd[k] = 3.0e38f;
    bi[k] = 0;
  }

  for (int mt = 0; mt < NT; ++mt) {
    const int mbase = mt * TILE;
    bf16_t(*Xm)[LDSPITCH] = sXm[mt & 1];
    float(*sDist)[65] = reinterpret_cast<float(*)[65]>(&Xm[0][0]);

    v8f a0 = {};
    v8f a1 = {};
#pragma unroll
    for (int ks = 0; ks < 6; ++ks) {  // K = 192 = 6 * 32
      const int kb = ks * 32;
      v16bf fa = frag_a(sXn, tr * 16 + nl, kb, hi);
      v16bf fb0 = frag_b(Xm, tc0 * 16 + nl, kb, hi);
      v16bf fb1 = frag_b(Xm, tc1 * 16 + nl, kb, hi);
      a0 = __builtin_amdgcn_wmma_f32_16x16x32_bf16(false, fa, false, fb0,
                                                   (short)0, a0, false, false);
      a1 = __builtin_amdgcn_wmma_f32_16x16x32_bf16(false, fa, false, fb1,
                                                   (short)0, a1, false, false);
    }
    __syncthreads();  // all waves done reading Xm -> reuse as dist tile

    const float s0 = sq[(size_t)b * NPTS + mbase + tc0 * 16 + nl];
    const float s1 = sq[(size_t)b * NPTS + mbase + tc1 * 16 + nl];
#pragma unroll
    for (int r = 0; r < 8; ++r) {  // C/D layout: M = r + 8*hi, N = nl
      const int nrow = tr * 16 + r + 8 * hi;
      sDist[nrow][tc0 * 16 + nl] = s0 - 2.0f * a0[r];
      sDist[nrow][tc1 * 16 + nl] = s1 - 2.0f * a1[r];
    }
    __syncthreads();

    if (t < TILE) {  // 64 scanner threads, register-resident sorted top-9
      for (int mm = 0; mm < TILE; ++mm) {
        const float d = sDist[t][mm];
        if (d < bd[8]) {
          bd[8] = d;
          bi[8] = mbase + mm;
#pragma unroll
          for (int j = 8; j > 0; --j) {
            if (bd[j] < bd[j - 1]) {
              float td = bd[j]; bd[j] = bd[j - 1]; bd[j - 1] = td;
              int ti = bi[j]; bi[j] = bi[j - 1]; bi[j - 1] = ti;
            }
          }
        }
      }
    }
    __syncthreads();  // scan done; current buffer free for refill

    if (mt + 1 < NT) {
      if (mt + 2 < NT) {
        // refill the just-freed buffer with tile mt+2, then wait such that
        // tile mt+1 (older, in-order) is guaranteed complete.
        async_fill_tile(sXm[mt & 1], xb + (size_t)(mbase + 2 * TILE) * C_IN, t);
        ASYNC_WAIT_6;
      } else {
        ASYNC_WAIT_0;  // tail: just drain tile mt+1
      }
      __syncthreads();
    }
  }

  if (t < TILE) {
    int* dst = nn + ((size_t)b * NPTS + nbase + t) * KNN;
#pragma unroll
    for (int k = 0; k < KNN; ++k) dst[k] = bi[k];
  }
}

// ---------------------------------------------------------------------------
// Kernel C: A1/A2 GEMMs: (25088 x 192) x (192 x 192), bf16 WMMA, f32 out.
// blockIdx.z selects (WA -> A1) vs (WB -> A2).
// ---------------------------------------------------------------------------
__global__ __launch_bounds__(256) void edge_gemm_kernel(
    const bf16_t* __restrict__ xth, const bf16_t* __restrict__ wa,
    const bf16_t* __restrict__ wb, float* __restrict__ out_a,
    float* __restrict__ out_b) {
  const bf16_t* w = blockIdx.z ? wb : wa;
  float* out = blockIdx.z ? out_b : out_a;
  const int rowbase = blockIdx.x * TILE;
  const int obase = blockIdx.y * TILE;
  const int t = threadIdx.x;

  __shared__ __align__(16) bf16_t sX[TILE][LDSPITCH];
  __shared__ __align__(16) bf16_t sW[TILE][LDSPITCH];

  async_fill_tile(sX, xth + (size_t)rowbase * C_IN, t);
  async_fill_tile(sW, w + (size_t)obase * C_IN, t);
  ASYNC_WAIT_0;
  __syncthreads();

  const int wv = t >> 5, lane = t & 31;
  const int hi = lane >> 4, nl = lane & 15;
  const int tr = wv & 3;
  const int tc0 = wv >> 2, tc1 = tc0 + 2;

  v8f a0 = {};
  v8f a1 = {};
#pragma unroll
  for (int ks = 0; ks < 6; ++ks) {
    const int kb = ks * 32;
    v16bf fa = frag_a(sX, tr * 16 + nl, kb, hi);
    v16bf fb0 = frag_b(sW, tc0 * 16 + nl, kb, hi);
    v16bf fb1 = frag_b(sW, tc1 * 16 + nl, kb, hi);
    a0 = __builtin_amdgcn_wmma_f32_16x16x32_bf16(false, fa, false, fb0,
                                                 (short)0, a0, false, false);
    a1 = __builtin_amdgcn_wmma_f32_16x16x32_bf16(false, fa, false, fb1,
                                                 (short)0, a1, false, false);
  }
#pragma unroll
  for (int r = 0; r < 8; ++r) {
    const int row = rowbase + tr * 16 + r + 8 * hi;
    out[(size_t)row * O_OUT + obase + tc0 * 16 + nl] = a0[r];
    out[(size_t)row * O_OUT + obase + tc1 * 16 + nl] = a1[r];
  }
}

// ---------------------------------------------------------------------------
// Kernel D: h = A1[n] + A2[nn[k]] + bias ; gelu(exact) ; max over K.
// LDS transpose so the write side is [b][o][n] (upsample-friendly).
// ---------------------------------------------------------------------------
__global__ __launch_bounds__(256) void combine_kernel(
    const float* __restrict__ a1, const float* __restrict__ a2,
    const int* __restrict__ nn, const float* __restrict__ bias,
    float* __restrict__ pooled) {
  const int b = blockIdx.y;
  const int nbase = blockIdx.x * TILE;
  const int t = threadIdx.x;

  __shared__ float sOut[O_OUT][TILE];  // 48KB
  __shared__ int sIdx[TILE][KNN];
  __shared__ float sBias[O_OUT];

  for (int idx = t; idx < TILE * KNN; idx += 256)
    sIdx[idx / KNN][idx % KNN] = nn[((size_t)b * NPTS + nbase) * KNN + idx];
  if (t < O_OUT) sBias[t] = bias[t];
  __syncthreads();

  for (int idx = t; idx < TILE * O_OUT; idx += 256) {
    const int nl = idx / O_OUT;
    const int o = idx % O_OUT;  // fastest -> coalesced A1/A2 reads
    const float base = a1[((size_t)b * NPTS + nbase + nl) * O_OUT + o] + sBias[o];
    float mx = -3.0e38f;
#pragma unroll
    for (int k = 0; k < KNN; ++k) {
      const int j = sIdx[nl][k];
      const float v = base + a2[((size_t)b * NPTS + j) * O_OUT + o];
      const float g = 0.5f * v * (1.0f + erff(v * 0.70710678118654752f));
      mx = fmaxf(mx, g);
    }
    sOut[o][nl] = mx;
  }
  __syncthreads();
  for (int idx = t; idx < TILE * O_OUT; idx += 256) {
    const int o = idx / TILE;
    const int nl = idx % TILE;  // fastest -> coalesced pooled writes
    pooled[((size_t)(b * O_OUT + o)) * NPTS + nbase + nl] = sOut[o][nl];
  }
}

// ---------------------------------------------------------------------------
// Kernel E: bilinear 2x upsample, half-pixel centers, edge clamp.
// ---------------------------------------------------------------------------
__global__ __launch_bounds__(256) void upsample_kernel(
    const float* __restrict__ pooled, float* __restrict__ out, int total) {
  const int gid = blockIdx.x * 256 + threadIdx.x;
  if (gid >= total) return;
  const int w = gid % HIN;
  const int h = (gid / HIN) % HIN;
  const int oc = (gid / (HIN * HIN)) % O_OUT;
  const int b = gid / (HIN * HIN * O_OUT);

  const float fy = h * 0.5f - 0.25f;
  const float fx = w * 0.5f - 0.25f;
  const float y0f = floorf(fy), x0f = floorf(fx);
  const float ty = fy - y0f, tx = fx - x0f;
  const int y0 = max(0, min(HD - 1, (int)y0f));
  const int y1 = max(0, min(HD - 1, (int)y0f + 1));
  const int x0 = max(0, min(HD - 1, (int)x0f));
  const int x1 = max(0, min(HD - 1, (int)x0f + 1));

  const float* p = pooled + ((size_t)(b * O_OUT + oc)) * NPTS;
  const float v00 = p[y0 * HD + x0], v01 = p[y0 * HD + x1];
  const float v10 = p[y1 * HD + x0], v11 = p[y1 * HD + x1];
  const float v0 = v00 + tx * (v01 - v00);
  const float v1 = v10 + tx * (v11 - v10);
  out[gid] = v0 + ty * (v1 - v0);
}

// ---------------------------------------------------------------------------
extern "C" void kernel_launch(void* const* d_in, const int* in_sizes, int n_in,
                              void* d_out, int out_size, void* d_ws,
                              size_t ws_size, hipStream_t stream) {
  const float* x = (const float*)d_in[0];       // (8,192,112,112)
  const float* w_conv = (const float*)d_in[1];  // (192,384)
  const float* b_conv = (const float*)d_in[2];  // (192,)
  float* out = (float*)d_out;

  // workspace carve-up (all 16B aligned)
  char* ws = (char*)d_ws;
  bf16_t* XTH = (bf16_t*)(ws + 0);        //  9,633,792 B
  float* SQ = (float*)(ws + 9633792);     //    100,352 B
  bf16_t* WAT = (bf16_t*)(ws + 9734144);  //     73,728 B
  bf16_t* WBT = (bf16_t*)(ws + 9807872);  //     73,728 B
  int* NN = (int*)(ws + 9881600);         //    903,168 B
  float* A1 = (float*)(ws + 10784768);    // 19,267,584 B
  float* A2 = (float*)(ws + 30052352);    // 19,267,584 B
  float* POOL = (float*)(ws + 49319936);  // 19,267,584 B
  // total ~65.4 MB

  avgpool_kernel<<<dim3(HD, BATCH), 256, 0, stream>>>(x, XTH, SQ);
  prep_w_kernel<<<dim3((O_OUT * C_IN + 255) / 256), 256, 0, stream>>>(w_conv,
                                                                      WAT, WBT);
  gram_topk_kernel<<<dim3(NT, BATCH), 256, 0, stream>>>(XTH, SQ, NN);
  edge_gemm_kernel<<<dim3(ROWS_TOT / TILE, O_OUT / TILE, 2), 256, 0, stream>>>(
      XTH, WAT, WBT, A1, A2);
  combine_kernel<<<dim3(NT, BATCH), 256, 0, stream>>>(A1, A2, NN, b_conv, POOL);
  const int total = BATCH * O_OUT * HIN * HIN;
  upsample_kernel<<<dim3(total / 256), 256, 0, stream>>>(POOL, out, total);
}
